// EncoderDecodeGNNForce_85487029060213
// MI455X (gfx1250) — matrix operand order repaired and checked by reading, not compiled, not run on hardware
//
#include <hip/hip_runtime.h>
#include <hip/hip_bf16.h>
#include <cstdint>
#include <cstddef>

typedef __attribute__((ext_vector_type(16))) _Float16 v16h;
typedef __attribute__((ext_vector_type(8)))  float    v8f;
typedef __attribute__((ext_vector_type(4)))  unsigned int u32x4;
typedef __attribute__((ext_vector_type(8)))  int      i32x8;
typedef __attribute__((ext_vector_type(4)))  int      i32x4;

#define DEVI __device__ __forceinline__

#if __has_builtin(__builtin_amdgcn_tensor_load_to_lds) && __has_builtin(__builtin_amdgcn_s_wait_tensorcnt)
#define USE_TDM 1
#else
#define USE_TDM 0
#endif

struct Seg { const void* row; int kts; int isf32; };

DEVI v8f zero8(){
  v8f z;
  #pragma unroll
  for (int i=0;i<8;i++) z[i]=0.f;
  return z;
}

// ---- WMMA fragment loaders -------------------------------------------------
// A (16x32 f16): lane holds row m=lane%16; k = kt*32 + {8h+0..7, 16+8h+0..7}
DEVI v16h load_a_h(const _Float16* row, int kt, int h){
  const _Float16* p = row + kt*32 + h*8;
  v16h r;
  #pragma unroll
  for (int i=0;i<8;i++) r[i] = p[i];
  #pragma unroll
  for (int i=0;i<8;i++) r[8+i] = p[16+i];
  return r;
}
DEVI v16h load_a_f(const float* row, int kt, int h){
  const float* p = row + kt*32 + h*8;
  v16h r;
  #pragma unroll
  for (int i=0;i<8;i++) r[i] = (_Float16)p[i];
  #pragma unroll
  for (int i=0;i<8;i++) r[8+i] = (_Float16)p[16+i];
  return r;
}
// B: pre-packed fragment-major: [kt][nt][lane][e], lane holds col n=lane%16,
// k = kt*32 + (e<8 ? 8*(lane>>4)+e : 16+8*(lane>>4)+e-8) -> one 32B load per lane.
DEVI v16h load_b(const _Float16* Wp, int kt, int nt, int ntiles){
  int lane = threadIdx.x & 31;
  return *(const v16h*)(Wp + ((size_t)(kt*ntiles + nt)*32 + lane)*16);
}
DEVI v8f wmma16(v16h a, v16h b, v8f c){
  return __builtin_amdgcn_wmma_f32_16x16x32_f16(false, a, false, b, (short)0, c, false, false);
}

#if USE_TDM
// TDM: async DMA of a packed weight array (ne f16 elements) global -> LDS.
// Encoded as a 1-row 2D tensor: data_size=2B, tensor_dim0=tile_dim0=ne,
// tensor_dim0_stride=ne, workgroup_mask=0 (not in cluster), D# type=2.
DEVI void tdm_load_1d(const void* gsrc, unsigned lds_off, unsigned ne){
  unsigned long long ga = (unsigned long long)(uintptr_t)gsrc;
  u32x4 g0;
  g0[0] = 1u;                                   // count=1 (valid user D#)
  g0[1] = lds_off;                              // lds_addr  (bits 63:32)
  g0[2] = (unsigned)ga;                         // global_addr[31:0]
  g0[3] = (unsigned)((ga >> 32) & 0x01FFFFFFu)  // global_addr[56:32]
        | 0x80000000u;                          // type=2 ("image")
  i32x8 g1;
  g1[0] = (int)0x00010000u;                     // data_size=1 (2B), mask=0
  g1[1] = (int)((ne & 0xFFFFu) << 16);          // tensor_dim0[15:0]
  g1[2] = (int)(((ne >> 16) & 0xFFFFu)          // tensor_dim0[31:16]
        | (1u << 16));                          // tensor_dim1 = 1
  g1[3] = (int)((ne & 0xFFFFu) << 16);          // tile_dim0 = ne
  g1[4] = 1;                                    // tile_dim1 = 1
  g1[5] = (int)ne;                              // tensor_dim0_stride[31:0]
  g1[6] = 0;
  g1[7] = 0;
  i32x4 z4; z4[0]=0; z4[1]=0; z4[2]=0; z4[3]=0;
#if defined(__clang_major__) && (__clang_major__ >= 23)
  i32x8 z8;
  #pragma unroll
  for (int i=0;i<8;i++) z8[i]=0;
  __builtin_amdgcn_tensor_load_to_lds(g0, g1, z4, z4, z8, 0);
#else
  __builtin_amdgcn_tensor_load_to_lds(g0, g1, z4, z4, 0);
#endif
}
#endif

// ---- generic fused MLP tile: out = relu(A@W1+b1) @ W2  (bias2 added by caller)
DEVI void mlp_l1(const Seg* segs, int nsegs, const _Float16* W1p, v8f* acc){
  int lane = threadIdx.x & 31, h = lane >> 4;
  int ktg = 0;
  for (int s=0;s<nsegs;s++){
    for (int i=0;i<segs[s].kts;i++){
      v16h a = segs[s].isf32 ? load_a_f((const float*)segs[s].row, i, h)
                             : load_a_h((const _Float16*)segs[s].row, i, h);
      #pragma unroll
      for (int t=0;t<8;t++){
        v16h b = load_b(W1p, ktg, t, 8);
        acc[t] = wmma16(a, b, acc[t]);
      }
      ktg++;
    }
  }
}

template<int OUT_TILES>
DEVI void mlp2_tile(const Seg* segs, int nsegs,
                    const _Float16* W1p, const float* b1, const _Float16* W2p,
                    _Float16* ldsh, v8f* out){
  int lane = threadIdx.x & 31, h = lane>>4, nl = lane&15;
  v8f acc[8];
  #pragma unroll
  for (int t=0;t<8;t++) acc[t]=zero8();
  mlp_l1(segs, nsegs, W1p, acc);
  // hidden: bias + relu -> LDS row-major (re-fragment D layout -> A layout)
  #pragma unroll
  for (int t=0;t<8;t++){
    float bb = b1[t*16+nl];
    #pragma unroll
    for (int v=0;v<8;v++){
      float xv = acc[t][v] + bb;
      ldsh[(v+8*h)*128 + t*16+nl] = (_Float16)(xv > 0.f ? xv : 0.f);
    }
  }
  __syncthreads();
  #pragma unroll
  for (int t=0;t<OUT_TILES;t++) out[t]=zero8();
  const _Float16* hrow = ldsh + nl*128;
  #pragma unroll
  for (int kt=0;kt<4;kt++){
    v16h a = load_a_h(hrow, kt, h);
    #pragma unroll
    for (int t=0;t<OUT_TILES;t++){
      v16h b = load_b(W2p, kt, t, OUT_TILES);
      out[t] = wmma16(a, b, out[t]);
    }
  }
}

// ---- weight packing: f32 (K x Nact) -> f16 fragment-major (Kpad x Npad) ----
__global__ void k_pack(const float* __restrict__ W, _Float16* __restrict__ dst,
                       int K, int Kpad, int Nact, int Npad){
  int ntiles = Npad >> 4;
  size_t per = (size_t)Kpad * Npad;
  W   += (size_t)blockIdx.y * K * Nact;
  dst += (size_t)blockIdx.y * per;
  for (size_t idx = blockIdx.x*(size_t)blockDim.x + threadIdx.x; idx < per;
       idx += gridDim.x*(size_t)blockDim.x){
    int e    = (int)(idx & 15);
    int lane = (int)((idx >> 4) & 31);
    size_t rest = idx >> 9;
    int nt = (int)(rest % ntiles);
    int kt = (int)(rest / ntiles);
    int h = lane >> 4, nl = lane & 15;
    int k = kt*32 + (e < 8 ? 8*h + e : 16 + 8*h + (e - 8));
    int col = nt*16 + nl;
    float v = (k < K && col < Nact) ? W[(size_t)k*Nact + col] : 0.f;
    dst[idx] = (_Float16)v;
  }
}

// ---- prep kernels ----------------------------------------------------------
__global__ void k_prep_xt(const float* x, const float* xinit, const float* mass,
                          _Float16* xt, int n){
  long i = blockIdx.x*(long)blockDim.x + threadIdx.x;
  if (i >= n) return;
  _Float16* r = xt + i*32;
  const float* xb = x + i*45;            // (9,T=5) row-major
  #pragma unroll
  for (int c=0;c<3;c++) r[c]   = (_Float16)xb[(6+c)*5 + 4];  // u_t
  #pragma unroll
  for (int c=0;c<3;c++) r[3+c] = (_Float16)xb[(3+c)*5 + 4];  // v_t
  #pragma unroll
  for (int c=0;c<3;c++) r[6+c] = (_Float16)xinit[i*3+c];
  r[9] = (_Float16)mass[i];
  #pragma unroll
  for (int j=10;j<32;j++) r[j] = (_Float16)0.f;
}

__global__ void k_prep_ea(const float* ea, _Float16* out, long E_){
  long i = blockIdx.x*(long)blockDim.x + threadIdx.x;
  if (i >= E_) return;
  _Float16* r = out + i*32;
  #pragma unroll
  for (int c=0;c<4;c++) r[c] = (_Float16)ea[i*4+c];
  #pragma unroll
  for (int j=4;j<32;j++) r[j] = (_Float16)0.f;
}

__global__ void k_zero(float* p, size_t n){
  size_t i = blockIdx.x*(size_t)blockDim.x + threadIdx.x;
  if (i < n) p[i] = 0.f;
}

__global__ void k_setmask(const int* idx, float* mask, long n){
  long i = blockIdx.x*(long)blockDim.x + threadIdx.x;
  if (i < n) mask[idx[i]] = 1.f;
}

__global__ void k_sagg(const float* hs, const int* ssrc, const int* sdst,
                       float* sagg, long ES_){
  long i = blockIdx.x*(long)blockDim.x + threadIdx.x;
  if (i >= ES_*128) return;
  long e = i >> 7; int c = (int)(i & 127);
  atomicAdd(&sagg[(size_t)sdst[e]*128 + c], hs[(size_t)ssrc[e]*128 + c]);
}

// ---- GEMM kernels ----------------------------------------------------------
__global__ __launch_bounds__(256) void k_encoder(
    const _Float16* xt, const _Float16* W1p, const float* b1,
    const _Float16* W2p, const float* b2, float* h32, _Float16* h16, int n){
  __shared__ _Float16 ldsh[8*2048];
  int wave = threadIdx.x>>5, lane = threadIdx.x&31, h = lane>>4, nl = lane&15;
  long tile = blockIdx.x*8L + wave;
  long row = tile*16 + nl; if (row >= n) row = n-1;
  Seg segs[1] = { { xt + row*32, 1, 0 } };
  v8f out[8];
  mlp2_tile<8>(segs, 1, W1p, b1, W2p, ldsh + wave*2048, out);
  #pragma unroll
  for (int t=0;t<8;t++){
    int col = t*16+nl; float bb = b2[col];
    #pragma unroll
    for (int v=0;v<8;v++){
      long rr = tile*16 + v + 8*h;
      if (rr < n){
        float val = out[t][v] + bb;
        h32[rr*128 + col] = val;
        h16[rr*128 + col] = (_Float16)val;
      }
    }
  }
}

__global__ __launch_bounds__(256) void k_edge_enc(
    const _Float16* ea, const _Float16* W1p, const float* b1,
    const _Float16* W2p, const float* b2, float* ef32, long E_){
  __shared__ _Float16 ldsh[8*2048];
  int wave = threadIdx.x>>5, lane = threadIdx.x&31, h = lane>>4, nl = lane&15;
  long tile = blockIdx.x*8L + wave;
  long row = tile*16 + nl; if (row >= E_) row = E_-1;
  Seg segs[1] = { { ea + row*32, 1, 0 } };
  v8f out[8];
  mlp2_tile<8>(segs, 1, W1p, b1, W2p, ldsh + wave*2048, out);
  #pragma unroll
  for (int t=0;t<8;t++){
    int col = t*16+nl; float bb = b2[col];
    #pragma unroll
    for (int v=0;v<8;v++){
      long rr = tile*16 + v + 8*h;
      if (rr < E_) ef32[rr*128 + col] = out[t][v] + bb;
    }
  }
}

// Hot kernel: per-edge MLP. Stage packed W1 (96KB) + W2 (32KB) in LDS once per
// block via TDM async DMA (TENSORcnt) when available, so WMMA B-fragments come
// from ds_load_b128 instead of re-streaming ~128KB/tile from L2.
// Total LDS = 160KB -> 2 blocks/WGP.
__global__ __launch_bounds__(256) void k_edge_layer(
    const _Float16* __restrict__ h16, float* __restrict__ ef32,
    const int* __restrict__ src, const int* __restrict__ dst,
    const _Float16* __restrict__ W1p, const float* __restrict__ b1,
    const _Float16* __restrict__ W2p, const float* __restrict__ b2,
    float* __restrict__ agg, long E_){
  __shared__ _Float16 w1s[384*128];   // 96 KB
  __shared__ _Float16 w2s[128*128];   // 32 KB
  __shared__ _Float16 ldsh[8*2048];   // 32 KB hidden
#if USE_TDM
  if (threadIdx.x == 0){
    tdm_load_1d(W1p, (unsigned)(uintptr_t)(void*)w1s, 384*128);
    tdm_load_1d(W2p, (unsigned)(uintptr_t)(void*)w2s, 128*128);
    __builtin_amdgcn_s_wait_tensorcnt(0);
  }
#else
  {
    const uint4* s1 = (const uint4*)W1p; uint4* d1 = (uint4*)w1s;
    for (int i = threadIdx.x; i < 384*128/8; i += 256) d1[i] = s1[i];
    const uint4* s2 = (const uint4*)W2p; uint4* d2 = (uint4*)w2s;
    for (int i = threadIdx.x; i < 128*128/8; i += 256) d2[i] = s2[i];
  }
#endif
  __syncthreads();

  int wave = threadIdx.x>>5, lane = threadIdx.x&31, h = lane>>4, nl = lane&15;
  long tile = blockIdx.x*8L + wave;
  long e = tile*16 + nl; if (e >= E_) e = E_-1;
  int se = src[e], de = dst[e];
  Seg segs[3] = {
    { h16  + (size_t)se*128, 4, 0 },
    { h16  + (size_t)de*128, 4, 0 },
    { ef32 + (size_t)e *128, 4, 1 },
  };
  v8f out[8];
  mlp2_tile<8>(segs, 3, w1s, b1, w2s, ldsh + wave*2048, out);
  int dsts[8]; long eos[8];
  #pragma unroll
  for (int v=0;v<8;v++){
    long eo = tile*16 + v + 8*h;
    eos[v] = eo;
    dsts[v] = dst[eo < E_ ? eo : E_-1];
  }
  #pragma unroll
  for (int t=0;t<8;t++){
    int col = t*16+nl; float bb = b2[col];
    #pragma unroll
    for (int v=0;v<8;v++){
      if (eos[v] < E_){
        float mval = out[t][v] + bb;
        size_t off = (size_t)eos[v]*128 + col;
        ef32[off] += mval;                                   // ef = ef + m
        atomicAdd(&agg[(size_t)dsts[v]*128 + col], mval);    // segment_sum
      }
    }
  }
}

__global__ __launch_bounds__(256) void k_node_layer(
    _Float16* h16, float* h32, const float* agg,
    const _Float16* W1p, const float* b1, const _Float16* W2p, const float* b2,
    int n){
  __shared__ _Float16 ldsh[8*2048];
  int wave = threadIdx.x>>5, lane = threadIdx.x&31, h = lane>>4, nl = lane&15;
  long tile = blockIdx.x*8L + wave;
  long row = tile*16 + nl; if (row >= n) row = n-1;
  Seg segs[2] = {
    { h16 + row*128, 4, 0 },
    { agg + row*128, 4, 1 },
  };
  v8f out[8];
  mlp2_tile<8>(segs, 2, W1p, b1, W2p, ldsh + wave*2048, out);
  #pragma unroll
  for (int t=0;t<8;t++){
    int col = t*16+nl; float bb = b2[col];
    #pragma unroll
    for (int v=0;v<8;v++){
      long rr = tile*16 + v + 8*h;
      if (rr < n){
        float val = h32[rr*128+col] + out[t][v] + bb;   // residual
        h32[rr*128+col] = val;
        h16[rr*128+col] = (_Float16)val;
      }
    }
  }
}

// 2-layer MLP head 128 -> 128 -> 3 (padded to 16)
__global__ __launch_bounds__(256) void k_force(
    const _Float16* in16, const _Float16* W1p, const float* b1,
    const _Float16* W2p, const float* b2s, float* fout, int n){
  __shared__ _Float16 ldsh[8*2048];
  int wave = threadIdx.x>>5, lane = threadIdx.x&31, h = lane>>4, nl = lane&15;
  long tile = blockIdx.x*8L + wave;
  long row = tile*16 + nl; if (row >= n) row = n-1;
  Seg segs[1] = { { in16 + row*128, 4, 0 } };
  v8f out[1];
  mlp2_tile<1>(segs, 1, W1p, b1, W2p, ldsh + wave*2048, out);
  #pragma unroll
  for (int v=0;v<8;v++){
    long rr = tile*16 + v + 8*h;
    if (rr < n && nl < 3) fout[rr*3 + nl] = out[0][v] + b2s[nl];
  }
}

// single layer: hs = relu(x_t @ s_W1 + b1)
__global__ __launch_bounds__(256) void k_hs(
    const _Float16* xt, const _Float16* W1p, const float* b1, float* hs, int n){
  int wave = threadIdx.x>>5, lane = threadIdx.x&31, h = lane>>4, nl = lane&15;
  long tile = blockIdx.x*8L + wave;
  long row = tile*16 + nl; if (row >= n) row = n-1;
  Seg segs[1] = { { xt + row*32, 1, 0 } };
  v8f acc[8];
  #pragma unroll
  for (int t=0;t<8;t++) acc[t]=zero8();
  mlp_l1(segs, 1, W1p, acc);
  #pragma unroll
  for (int t=0;t<8;t++){
    int col = t*16+nl; float bb = b1[col];
    #pragma unroll
    for (int v=0;v<8;v++){
      long rr = tile*16 + v + 8*h;
      if (rr < n){
        float xv = acc[t][v] + bb;
        hs[rr*128+col] = xv > 0.f ? xv : 0.f;
      }
    }
  }
}

// single layer: h_surf = relu([hs, sagg] @ s_W2 + b2) -> f16
__global__ __launch_bounds__(256) void k_hsurf(
    const float* hs, const float* sagg, const _Float16* Wp, const float* b,
    _Float16* hsurf16, int n){
  int wave = threadIdx.x>>5, lane = threadIdx.x&31, h = lane>>4, nl = lane&15;
  long tile = blockIdx.x*8L + wave;
  long row = tile*16 + nl; if (row >= n) row = n-1;
  Seg segs[2] = { { hs + row*128, 4, 1 }, { sagg + row*128, 4, 1 } };
  v8f acc[8];
  #pragma unroll
  for (int t=0;t<8;t++) acc[t]=zero8();
  mlp_l1(segs, 2, Wp, acc);
  #pragma unroll
  for (int t=0;t<8;t++){
    int col = t*16+nl; float bb = b[col];
    #pragma unroll
    for (int v=0;v<8;v++){
      long rr = tile*16 + v + 8*h;
      if (rr < n){
        float xv = acc[t][v] + bb;
        hsurf16[rr*128+col] = (_Float16)(xv > 0.f ? xv : 0.f);
      }
    }
  }
}

// a_pred + dec_in construction + loss term 1
__global__ void k_apred(const float* fint, const float* fext, const float* mask,
                        const float* x, const float* dt, float* apred,
                        _Float16* dec16, float* lossacc, int n){
  long i = blockIdx.x*(long)blockDim.x + threadIdx.x;
  if (i >= n) return;
  float mk = mask[i], s = 0.f, ap[3];
  #pragma unroll
  for (int c=0;c<3;c++){
    ap[c] = fint[i*3+c] + fext[i*3+c]*mk;
    apred[i*3+c] = ap[c];
    float at = x[(i*9 + c)*5 + 4];
    float d = ap[c] - at; s += d*d;
  }
  atomicAdd(lossacc, s);
  _Float16* r = dec16 + i*64;
  for (int j=0;j<45;j++){
    int c = j/5, t = j%5;
    float v = (c<3 && t==4) ? ap[c] : x[i*45 + j];
    r[j] = (_Float16)v;
  }
  r[45] = (_Float16)dt[i];
  #pragma unroll
  for (int j=46;j<64;j++) r[j] = (_Float16)0.f;
}

// decoder mlp + integrator + y_t store + loss term 2
__global__ __launch_bounds__(256) void k_decoder(
    const _Float16* dec16, const _Float16* W1p, const float* b1,
    const _Float16* W2p, const float* b2, const float* x, const float* y,
    const float* dt, const float* apred, float* yt_out, float* lossacc, int n){
  __shared__ _Float16 ldsh[8*2048];
  int wave = threadIdx.x>>5, lane = threadIdx.x&31, h = lane>>4, nl = lane&15;
  long tile = blockIdx.x*8L + wave;
  long row = tile*16 + nl; if (row >= n) row = n-1;
  Seg segs[1] = { { dec16 + row*64, 2, 0 } };
  v8f out[1];
  mlp2_tile<1>(segs, 1, W1p, b1, W2p, ldsh + wave*2048, out);
  float lsum = 0.f;
  if (nl < 6){
    #pragma unroll
    for (int v=0;v<8;v++){
      long rr = tile*16 + v + 8*h;
      if (rr < n){
        float dv  = out[0][v] + b2[nl];
        float dtv = dt[rr];
        float yt;
        if (nl < 3){
          float vt = x[(rr*9 + 3 + nl)*5 + 4];
          float ap = apred[rr*3 + nl];
          yt = vt + ap*dtv + dv*dtv;
        } else {
          int c2 = nl - 3;
          float ut = x[(rr*9 + 6 + c2)*5 + 4];
          float vt = x[(rr*9 + 3 + c2)*5 + 4];
          float ap = apred[rr*3 + c2];
          yt = ut + vt*dtv + 0.5f*ap*dtv*dtv + dv*dtv;
        }
        yt_out[rr*6 + nl] = yt;
        float d = yt - y[rr*9 + 3 + nl];
        lsum += d*d;
      }
    }
    atomicAdd(lossacc + 1, lsum);
  }
}

__global__ void k_loss_fin(const float* lossacc, float* out, int n){
  if (threadIdx.x == 0 && blockIdx.x == 0)
    out[0] = lossacc[0]/(3.0f*n) + lossacc[1]/(6.0f*n);
}

// ---- host side -------------------------------------------------------------
extern "C" void kernel_launch(void* const* d_in, const int* in_sizes, int n_in,
                              void* d_out, int out_size, void* d_ws, size_t ws_size,
                              hipStream_t stream) {
  const float* x     = (const float*)d_in[0];
  const float* xinit = (const float*)d_in[1];
  const float* mass  = (const float*)d_in[2];
  const float* eattr = (const float*)d_in[3];
  const float* dt    = (const float*)d_in[4];
  const float* y     = (const float*)d_in[5];
  const int*   eidx  = (const int*)d_in[6];
  const int*   esidx = (const int*)d_in[7];
  const float* enc_W1=(const float*)d_in[8],  *enc_b1=(const float*)d_in[9];
  const float* enc_W2=(const float*)d_in[10], *enc_b2=(const float*)d_in[11];
  const float* ee_W1 =(const float*)d_in[12], *ee_b1 =(const float*)d_in[13];
  const float* ee_W2 =(const float*)d_in[14], *ee_b2 =(const float*)d_in[15];
  const float* le_W1 =(const float*)d_in[16], *le_b1 =(const float*)d_in[17];
  const float* le_W2 =(const float*)d_in[18], *le_b2 =(const float*)d_in[19];
  const float* ln_W1 =(const float*)d_in[20], *ln_b1 =(const float*)d_in[21];
  const float* ln_W2 =(const float*)d_in[22], *ln_b2 =(const float*)d_in[23];
  const float* ht_W1 =(const float*)d_in[24], *ht_b1 =(const float*)d_in[25];
  const float* ht_W2 =(const float*)d_in[26], *ht_b2 =(const float*)d_in[27];
  const float* s_W1  =(const float*)d_in[28], *s_b1  =(const float*)d_in[29];
  const float* s_W2  =(const float*)d_in[30], *s_b2  =(const float*)d_in[31];
  const float* hs_W1 =(const float*)d_in[32], *hs_b1 =(const float*)d_in[33];
  const float* hs_W2 =(const float*)d_in[34], *hs_b2 =(const float*)d_in[35];
  const float* dec_W1=(const float*)d_in[36], *dec_b1=(const float*)d_in[37];
  const float* dec_W2=(const float*)d_in[38], *dec_b2=(const float*)d_in[39];

  const int  N  = in_sizes[2];
  const long E  = in_sizes[3] / 4;
  const long ES = in_sizes[7] / 2;
  const int  L_ = 3;

  const int* src  = eidx;          const int* dst  = eidx + E;
  const int* ssrc = esidx;         const int* sdst = esidx + ES;

  uintptr_t cur = (uintptr_t)d_ws;
  auto alloc = [&](size_t bytes) -> void* {
    uintptr_t r = (cur + 255) & ~(uintptr_t)255;
    cur = r + bytes;
    return (void*)r;
  };

  float*     h32    = (float*)    alloc((size_t)N*128*4);
  _Float16*  h16    = (_Float16*) alloc((size_t)N*128*2);
  float*     ef32   = (float*)    alloc((size_t)E*128*4);
  float*     agg    = (float*)    alloc((size_t)N*128*4);
  _Float16*  xt16   = (_Float16*) alloc((size_t)N*32*2);
  _Float16*  ea16   = (_Float16*) alloc((size_t)E*32*2);
  float*     hs32   = (float*)    alloc((size_t)N*128*4);
  float*     sagg   = (float*)    alloc((size_t)N*128*4);
  _Float16*  hsurf16= (_Float16*) alloc((size_t)N*128*2);
  float*     fint   = (float*)    alloc((size_t)N*3*4);
  float*     fext   = (float*)    alloc((size_t)N*3*4);
  float*     apred  = (float*)    alloc((size_t)N*3*4);
  float*     maskb  = (float*)    alloc((size_t)N*4);
  _Float16*  dec16  = (_Float16*) alloc((size_t)N*64*2);
  float*     lossacc= (float*)    alloc(2*4);

  _Float16* encW1p = (_Float16*)alloc(32*128*2);
  _Float16* encW2p = (_Float16*)alloc(128*128*2);
  _Float16* eeW1p  = (_Float16*)alloc(32*128*2);
  _Float16* eeW2p  = (_Float16*)alloc(128*128*2);
  _Float16* leW1p  = (_Float16*)alloc((size_t)L_*384*128*2);
  _Float16* leW2p  = (_Float16*)alloc((size_t)L_*128*128*2);
  _Float16* lnW1p  = (_Float16*)alloc((size_t)L_*256*128*2);
  _Float16* lnW2p  = (_Float16*)alloc((size_t)L_*128*128*2);
  _Float16* htW1p  = (_Float16*)alloc(128*128*2);
  _Float16* htW2p  = (_Float16*)alloc(128*16*2);
  _Float16* sW1p   = (_Float16*)alloc(32*128*2);
  _Float16* sW2p   = (_Float16*)alloc(256*128*2);
  _Float16* hsW1p  = (_Float16*)alloc(128*128*2);
  _Float16* hsW2p  = (_Float16*)alloc(128*16*2);
  _Float16* decW1p = (_Float16*)alloc(64*128*2);
  _Float16* decW2p = (_Float16*)alloc(128*16*2);
  (void)ws_size; (void)n_in; (void)out_size;

  const int TPB = 256;
  // pack weights (K, Kpad, Nact, Npad), gridDim.y = layers
  k_pack<<<dim3(16,1),TPB,0,stream>>>(enc_W1, encW1p, 10, 32, 128, 128);
  k_pack<<<dim3(32,1),TPB,0,stream>>>(enc_W2, encW2p, 128,128, 128, 128);
  k_pack<<<dim3(16,1),TPB,0,stream>>>(ee_W1,  eeW1p,  4,  32, 128, 128);
  k_pack<<<dim3(32,1),TPB,0,stream>>>(ee_W2,  eeW2p,  128,128, 128, 128);
  k_pack<<<dim3(64,L_),TPB,0,stream>>>(le_W1, leW1p,  384,384, 128, 128);
  k_pack<<<dim3(32,L_),TPB,0,stream>>>(le_W2, leW2p,  128,128, 128, 128);
  k_pack<<<dim3(48,L_),TPB,0,stream>>>(ln_W1, lnW1p,  256,256, 128, 128);
  k_pack<<<dim3(32,L_),TPB,0,stream>>>(ln_W2, lnW2p,  128,128, 128, 128);
  k_pack<<<dim3(32,1),TPB,0,stream>>>(ht_W1,  htW1p,  128,128, 128, 128);
  k_pack<<<dim3(8,1), TPB,0,stream>>>(ht_W2,  htW2p,  128,128, 3,   16);
  k_pack<<<dim3(16,1),TPB,0,stream>>>(s_W1,   sW1p,   10, 32, 128, 128);
  k_pack<<<dim3(48,1),TPB,0,stream>>>(s_W2,   sW2p,   256,256, 128, 128);
  k_pack<<<dim3(32,1),TPB,0,stream>>>(hs_W1,  hsW1p,  128,128, 128, 128);
  k_pack<<<dim3(8,1), TPB,0,stream>>>(hs_W2,  hsW2p,  128,128, 3,   16);
  k_pack<<<dim3(16,1),TPB,0,stream>>>(dec_W1, decW1p, 46, 64, 128, 128);
  k_pack<<<dim3(8,1), TPB,0,stream>>>(dec_W2, decW2p, 128,128, 6,   16);

  // prep
  k_prep_xt<<<(N+TPB-1)/TPB, TPB, 0, stream>>>(x, xinit, mass, xt16, N);
  k_prep_ea<<<(int)((E+TPB-1)/TPB), TPB, 0, stream>>>(eattr, ea16, E);

  const int nbN = (N + 127) / 128;
  const int nbE = (int)((E + 127) / 128);

  // encoders
  k_encoder <<<nbN, TPB, 0, stream>>>(xt16, encW1p, enc_b1, encW2p, enc_b2, h32, h16, N);
  k_edge_enc<<<nbE, TPB, 0, stream>>>(ea16, eeW1p, ee_b1, eeW2p, ee_b2, ef32, E);

  // message-passing layers
  for (int l=0; l<L_; ++l){
    k_zero<<<(int)(((size_t)N*128+TPB-1)/TPB), TPB, 0, stream>>>(agg, (size_t)N*128);
    k_edge_layer<<<nbE, TPB, 0, stream>>>(h16, ef32, src, dst,
        leW1p + (size_t)l*384*128, le_b1 + l*128,
        leW2p + (size_t)l*128*128, le_b2 + l*128, agg, E);
    k_node_layer<<<nbN, TPB, 0, stream>>>(h16, h32, agg,
        lnW1p + (size_t)l*256*128, ln_b1 + l*128,
        lnW2p + (size_t)l*128*128, ln_b2 + l*128, N);
  }

  // heads
  k_force<<<nbN, TPB, 0, stream>>>(h16, htW1p, ht_b1, htW2p, ht_b2, fint, N);

  k_hs<<<nbN, TPB, 0, stream>>>(xt16, sW1p, s_b1, hs32, N);
  k_zero<<<(int)(((size_t)N*128+TPB-1)/TPB), TPB, 0, stream>>>(sagg, (size_t)N*128);
  k_sagg<<<(int)((ES*128+TPB-1)/TPB), TPB, 0, stream>>>(hs32, ssrc, sdst, sagg, ES);
  k_hsurf<<<nbN, TPB, 0, stream>>>(hs32, sagg, sW2p, s_b2, hsurf16, N);
  k_force<<<nbN, TPB, 0, stream>>>(hsurf16, hsW1p, hs_b1, hsW2p, hs_b2, fext, N);

  k_zero<<<(N+TPB-1)/TPB, TPB, 0, stream>>>(maskb, (size_t)N);
  k_zero<<<1, TPB, 0, stream>>>(lossacc, 2);
  k_setmask<<<(int)((2*ES+TPB-1)/TPB), TPB, 0, stream>>>(esidx, maskb, 2*ES);

  k_apred<<<(N+TPB-1)/TPB, TPB, 0, stream>>>(fint, fext, maskb, x, dt, apred,
                                             dec16, lossacc, N);
  k_decoder<<<nbN, TPB, 0, stream>>>(dec16, decW1p, dec_b1, decW2p, dec_b2,
                                     x, y, dt, apred, (float*)d_out, lossacc, N);
  k_loss_fin<<<1, 1, 0, stream>>>(lossacc, (float*)d_out + (size_t)N*6, N);
}
